// ChaosSSMHybridBlock_29480655520347
// MI455X (gfx1250) — matrix operand order repaired and checked by reading, not compiled, not run on hardware
//
#include <hip/hip_runtime.h>

// ---------------------------------------------------------------------------
// ChaosSSMHybridBlock for MI455X (gfx1250, wave32, WMMA bf16)
// B=4, L=4096, D=1024, A=64, H=1, W=64, FF=2048  (reference dims, fixed)
// ---------------------------------------------------------------------------

typedef __attribute__((ext_vector_type(16))) __bf16 v16bf;
typedef __attribute__((ext_vector_type(8)))  float  v8f;

#define BB 4
#define LL 4096
#define DD 1024
#define AA 64
#define WW 64
#define FFH 2048
#define BL (BB * LL)          // 16384 rows
#define NCH 8                 // scan chunks
#define CLEN (LL / NCH)       // 512 steps per chunk

__device__ __forceinline__ __bf16 f2bf(float x) { return (__bf16)x; }
__device__ __forceinline__ float sigf(float x) { return 1.f / (1.f + __expf(-x)); }

// ---------------------------------------------------------------------------
// Weight prep: f32 [R,C] row-major  ->  bf16 [C,R] (i.e. [N,K] for GEMM B^T)
// ---------------------------------------------------------------------------
__global__ void k_convert_transpose(const float* __restrict__ in, __bf16* __restrict__ out,
                                    int R, int C) {
  int g = blockIdx.x * 256 + threadIdx.x;
  if (g >= R * C) return;
  int r = g / C, c = g % C;
  out[(size_t)c * R + r] = f2bf(in[(size_t)r * C + c]);
}

// ---------------------------------------------------------------------------
// u = rmsnorm(x, norm1_w) * b_w          one block (256 thr) per row
// ---------------------------------------------------------------------------
__global__ __launch_bounds__(256) void k_rmsnorm_mul(const float* __restrict__ x,
                                                     const float* __restrict__ w,
                                                     const float* __restrict__ bw,
                                                     float* __restrict__ u) {
  int row = blockIdx.x;
  const float* xr = x + (size_t)row * DD;
  float s = 0.f;
  for (int e = threadIdx.x; e < DD; e += 256) { float v = xr[e]; s += v * v; }
  for (int off = 16; off; off >>= 1) s += __shfl_xor(s, off, 32);
  __shared__ float red[8];
  if ((threadIdx.x & 31) == 0) red[threadIdx.x >> 5] = s;
  __syncthreads();
  float tot = 0.f;
#pragma unroll
  for (int i = 0; i < 8; ++i) tot += red[i];
  float r = rsqrtf(tot * (1.f / DD) + 1e-6f);
  for (int e = threadIdx.x; e < DD; e += 256)
    u[(size_t)row * DD + e] = xr[e] * r * w[e] * bw[e];
}

// n2 = rmsnorm(x2, norm2_w) -> bf16 (GEMM A operand)
__global__ __launch_bounds__(256) void k_rmsnorm_bf16(const float* __restrict__ x,
                                                      const float* __restrict__ w,
                                                      __bf16* __restrict__ out) {
  int row = blockIdx.x;
  const float* xr = x + (size_t)row * DD;
  float s = 0.f;
  for (int e = threadIdx.x; e < DD; e += 256) { float v = xr[e]; s += v * v; }
  for (int off = 16; off; off >>= 1) s += __shfl_xor(s, off, 32);
  __shared__ float red[8];
  if ((threadIdx.x & 31) == 0) red[threadIdx.x >> 5] = s;
  __syncthreads();
  float tot = 0.f;
#pragma unroll
  for (int i = 0; i < 8; ++i) tot += red[i];
  float r = rsqrtf(tot * (1.f / DD) + 1e-6f);
  for (int e = threadIdx.x; e < DD; e += 256)
    out[(size_t)row * DD + e] = f2bf(xr[e] * r * w[e]);
}

// ---------------------------------------------------------------------------
// Chunked diagonal SSM scan  h_t = a*h_{t-1} + u_t   (a = sigmoid(a_raw))
// pass 1: per-chunk local scans (h0 = 0), store h_local and chunk finals
// ---------------------------------------------------------------------------
__global__ __launch_bounds__(256) void k_scan_local(const float* __restrict__ u,
                                                    const float* __restrict__ a_raw,
                                                    float* __restrict__ hbuf,
                                                    float* __restrict__ cfin) {
  const int DB = DD / 256;
  int dblk = blockIdx.x % DB;
  int ch   = (blockIdx.x / DB) % NCH;
  int b    = blockIdx.x / (DB * NCH);
  int d = dblk * 256 + threadIdx.x;
  float a = sigf(a_raw[d]);
  size_t base = ((size_t)b * LL + (size_t)ch * CLEN) * DD + d;
  float h = 0.f;
  for (int t = 0; t < CLEN; ++t) {
    size_t idx = base + (size_t)t * DD;
    h = a * h + u[idx];
    hbuf[idx] = h;
  }
  cfin[((size_t)b * NCH + ch) * DD + d] = h;
}

// pass 2: carry prefix across the 8 chunks; carry factor per chunk = a^CLEN
__global__ void k_scan_carry(const float* __restrict__ a_raw,
                             const float* __restrict__ cfin,
                             float* __restrict__ cin) {
  int g = blockIdx.x * 256 + threadIdx.x;   // B*D threads
  int d = g % DD, b = g / DD;
  float a = sigf(a_raw[d]);
  float ap = a;
#pragma unroll
  for (int i = 0; i < 9; ++i) ap *= ap;     // a^512  (CLEN = 512 = 2^9)
  float c = 0.f;
#pragma unroll
  for (int ch = 0; ch < NCH; ++ch) {
    size_t idx = ((size_t)b * NCH + ch) * DD + d;
    cin[idx] = c;
    c = ap * c + cfin[idx];
  }
}

// pass 3: h = h_local + a^(off+1)*carry (incremental pow), fuse residual
//         x_ssm = x + h*c_w  (overwrites hbuf in place) + bf16 copy
__global__ __launch_bounds__(256) void k_scan_fixup(const float* __restrict__ x,
                                                    const float* __restrict__ a_raw,
                                                    const float* __restrict__ c_w,
                                                    const float* __restrict__ cin,
                                                    float* __restrict__ hbuf,
                                                    __bf16* __restrict__ xs_bf) {
  const int DB = DD / 256;
  int dblk = blockIdx.x % DB;
  int ch   = (blockIdx.x / DB) % NCH;
  int b    = blockIdx.x / (DB * NCH);
  int d = dblk * 256 + threadIdx.x;
  float a = sigf(a_raw[d]);
  float carry = cin[((size_t)b * NCH + ch) * DD + d];
  float cw = c_w[d];
  float pw = a;
  size_t base = ((size_t)b * LL + (size_t)ch * CLEN) * DD + d;
  for (int t = 0; t < CLEN; ++t) {
    size_t idx = base + (size_t)t * DD;
    float h = hbuf[idx] + pw * carry;
    pw *= a;
    float xs = x[idx] + h * cw;
    hbuf[idx] = xs;
    xs_bf[idx] = f2bf(xs);
  }
}

// ---------------------------------------------------------------------------
// gate[m] = sigmoid(x_ssm[m,:] . Wg + gate_bias)     one wave per row
// ---------------------------------------------------------------------------
__global__ __launch_bounds__(256) void k_gate(const float* __restrict__ xs,
                                              const float* __restrict__ wg,
                                              const float* __restrict__ gb,
                                              float* __restrict__ gate) {
  int m = blockIdx.x * 8 + (threadIdx.x >> 5);
  int lane = threadIdx.x & 31;
  const float* xr = xs + (size_t)m * DD;
  float s = 0.f;
  for (int e = lane; e < DD; e += 32) s += xr[e] * wg[e];
  for (int off = 16; off; off >>= 1) s += __shfl_xor(s, off, 32);
  if (lane == 0) gate[m] = sigf(s + gb[0]);
}

// ---------------------------------------------------------------------------
// Fragment loaders (16-bit A 16x32 / B 32x16; this code's contiguous k order, not the ISA table order).
// A row-major [M,K]: lane l -> row = l&15; k = 16h + e (contiguous 16).
// BT row-major [N,K]: lane l -> col = l&15; k = 16h + e (contiguous 16).
// ---------------------------------------------------------------------------
__device__ __forceinline__ v16bf load_afrag(const __bf16* __restrict__ ap) {
  v16bf f;
#pragma unroll
  for (int j = 0; j < 4; ++j) {
    f[2 * j]     = ap[2 * j];
    f[2 * j + 1] = ap[2 * j + 1];
    f[8 + 2 * j] = ap[8 + 2 * j];
    f[9 + 2 * j] = ap[8 + 2 * j + 1];
  }
  return f;
}
__device__ __forceinline__ v16bf load_bfrag(const __bf16* __restrict__ bp) {
  v16bf f;
#pragma unroll
  for (int e = 0; e < 16; ++e) f[e] = bp[e];
  return f;
}

// ---------------------------------------------------------------------------
// bf16 WMMA GEMM:  C[M,N] = A[M,K] x B[K,N], B pre-transposed to BT[N,K].
// 256 thr = 8 wave32 in a fixed 4(M) x 2(N) wave grid. Each wave owns a
// 32 x (16*WN) macro-tile: 2 A-frags + WN B-frags feed 2*WN v_wmma per k-step.
//   WN=2: block tile 128x64  (QKV, N=192)   -> 8 dwords VMEM / wmma
//   WN=4: block tile 128x128 (Wo/FFN)       -> 6 dwords VMEM / wmma
// EPI: 0 = raw f32   1 = silu -> bf16   2 = resid + gate*acc   3 = resid + acc
// ---------------------------------------------------------------------------
template <int EPI, int WN>
__global__ __launch_bounds__(256) void k_gemm_bf16(const __bf16* __restrict__ A,
                                                   const __bf16* __restrict__ BT,
                                                   float* outF, __bf16* outH,
                                                   const float* resid,
                                                   const float* gate,
                                                   int N, int K) {
  const int lane = threadIdx.x & 31;
  const int wave = threadIdx.x >> 5;
  const int tileM = blockIdx.y * 128 + (wave >> 1) * 32;
  const int tileN = blockIdx.x * (32 * WN) + (wave & 1) * (16 * WN);
  const int half = lane >> 4;
  const int l15  = lane & 15;

  const __bf16* arow0 = A + (size_t)(tileM + l15) * K + half * 16;
  const __bf16* arow1 = arow0 + (size_t)16 * K;
  const __bf16* brow[WN];
#pragma unroll
  for (int j = 0; j < WN; ++j)
    brow[j] = BT + (size_t)(tileN + j * 16 + l15) * K + half * 16;

  v8f acc[2][WN];
#pragma unroll
  for (int j = 0; j < WN; ++j) { acc[0][j] = (v8f){}; acc[1][j] = (v8f){}; }

  for (int kc = 0; kc < K; kc += 32) {
    v16bf a0 = load_afrag(arow0 + kc);
    v16bf a1 = load_afrag(arow1 + kc);
    v16bf bf[WN];
#pragma unroll
    for (int j = 0; j < WN; ++j) bf[j] = load_bfrag(brow[j] + kc);
#pragma unroll
    for (int j = 0; j < WN; ++j) {
      acc[0][j] = __builtin_amdgcn_wmma_f32_16x16x32_bf16(false, a0, false, bf[j],
                                                          (short)0, acc[0][j], false, false);
      acc[1][j] = __builtin_amdgcn_wmma_f32_16x16x32_bf16(false, a1, false, bf[j],
                                                          (short)0, acc[1][j], false, false);
    }
  }

  // epilogue: C/D layout -> vgpr i holds (M = 8*half + i, N = lane&15)
#pragma unroll
  for (int mi = 0; mi < 2; ++mi) {
#pragma unroll
    for (int j = 0; j < WN; ++j) {
      const int tm = tileM + mi * 16;
      const int tn = tileN + j * 16;
#pragma unroll
      for (int i = 0; i < 8; ++i) {
        int gm = tm + half * 8 + i;
        size_t idx = (size_t)gm * N + tn + l15;
        float v = acc[mi][j][i];
        if (EPI == 0) {
          outF[idx] = v;
        } else if (EPI == 1) {
          outH[idx] = f2bf(v * sigf(v));                 // silu
        } else if (EPI == 2) {
          outF[idx] = resid[idx] + gate[gm] * v;         // x2 = x_ssm + gate*attn
        } else {
          outF[idx] = resid[idx] + v;                    // y = x2 + ffn
        }
      }
    }
  }
}

// ---------------------------------------------------------------------------
// Causal sliding-window attention (W=64, hd=64, H=1). One wave per query.
// qkv packed per row: [0:64)=Q  [64:128)=K  [128:192)=V   (f32)
// ---------------------------------------------------------------------------
__global__ __launch_bounds__(256) void k_swa(const float* __restrict__ qkv,
                                             float* __restrict__ attn,
                                             __bf16* __restrict__ attnbf) {
  const int lane = threadIdx.x & 31;
  const int ws   = threadIdx.x >> 5;
  const int m = blockIdx.x * 8 + ws;
  const int b = m / LL, l = m % LL;
  const int base = b * LL;

  __shared__ float sq[8][64];
  __shared__ float sp[8][64];

  const float* qr = qkv + (size_t)m * 192;
  sq[ws][lane]      = qr[lane];
  sq[ws][lane + 32] = qr[lane + 32];
  __syncthreads();

  // two window slots per lane: w0 = lane, w1 = lane + 32
  float s0 = -1e30f, s1 = -1e30f;
  {
    int kp = l - (WW - 1) + lane;
    if (kp >= 0) {
      const float* kr = qkv + (size_t)(base + kp) * 192 + 64;
      float s = 0.f;
#pragma unroll 8
      for (int e = 0; e < 64; ++e) s += sq[ws][e] * kr[e];
      s0 = s * 0.125f;                 // 1/sqrt(64)
    }
  }
  {
    int kp = l - (WW - 1) + lane + 32;
    if (kp >= 0) {
      const float* kr = qkv + (size_t)(base + kp) * 192 + 64;
      float s = 0.f;
#pragma unroll 8
      for (int e = 0; e < 64; ++e) s += sq[ws][e] * kr[e];
      s1 = s * 0.125f;
    }
  }
  float mx = fmaxf(s0, s1);
  for (int off = 16; off; off >>= 1) mx = fmaxf(mx, __shfl_xor(mx, off, 32));
  float e0 = (s0 > -1e29f) ? __expf(s0 - mx) : 0.f;
  float e1 = (s1 > -1e29f) ? __expf(s1 - mx) : 0.f;
  float sum = e0 + e1;
  for (int off = 16; off; off >>= 1) sum += __shfl_xor(sum, off, 32);
  float inv = 1.f / sum;
  sp[ws][lane]      = e0 * inv;
  sp[ws][lane + 32] = e1 * inv;
  __syncthreads();

  // out[d] = sum_w p_w * V[kp(w)][d]; lane owns d = lane and lane+32
  float a0 = 0.f, a1 = 0.f;
  int kp0 = l - (WW - 1);
  int wstart = (kp0 < 0) ? -kp0 : 0;
  for (int w = wstart; w < WW; ++w) {
    const float* vr = qkv + (size_t)(base + kp0 + w) * 192 + 128;
    float p = sp[ws][w];
    a0 += p * vr[lane];
    a1 += p * vr[lane + 32];
  }
  size_t o = (size_t)m * 64;
  attn[o + lane]      = a0;
  attn[o + lane + 32] = a1;
  attnbf[o + lane]      = f2bf(a0);
  attnbf[o + lane + 32] = f2bf(a1);
}

// ---------------------------------------------------------------------------
// Launch
// ---------------------------------------------------------------------------
extern "C" void kernel_launch(void* const* d_in, const int* in_sizes, int n_in,
                              void* d_out, int out_size, void* d_ws, size_t ws_size,
                              hipStream_t stream) {
  const float* x       = (const float*)d_in[0];
  const float* norm1_w = (const float*)d_in[1];
  const float* norm2_w = (const float*)d_in[2];
  const float* a_raw   = (const float*)d_in[3];
  const float* b_w     = (const float*)d_in[4];
  const float* c_w     = (const float*)d_in[5];
  const float* Wq      = (const float*)d_in[6];
  const float* Wk      = (const float*)d_in[7];
  const float* Wv      = (const float*)d_in[8];
  const float* Wo      = (const float*)d_in[9];
  const float* Wg      = (const float*)d_in[10];
  const float* gbias   = (const float*)d_in[11];
  const float* ff_w1   = (const float*)d_in[12];
  const float* ff_w2   = (const float*)d_in[13];
  float* out = (float*)d_out;

  char* ws = (char*)d_ws;
  const size_t MB = 1ull << 20;
  float*  bufU    = (float*)(ws + 0);            // [BL,D] f32: u, then reused as bf16 FFN hidden
  float*  bufX    = (float*)(ws + 64 * MB);      // [BL,D] f32: h_local -> x_ssm -> x2 (in place)
  __bf16* bufXbf  = (__bf16*)(ws + 128 * MB);    // [BL,D] bf16: x_ssm_bf16, then n2_bf16
  float*  qkv     = (float*)(ws + 160 * MB);     // [BL,192] f32
  float*  attn    = (float*)(ws + 172 * MB);     // [BL,64] f32
  __bf16* attnbf  = (__bf16*)(ws + 176 * MB);    // [BL,64] bf16
  float*  gate    = (float*)(ws + 178 * MB);     // [BL] f32
  float*  cfin    = (float*)(ws + 179 * MB);     // [B,NCH,D]
  float*  cin     = (float*)(ws + 180 * MB);     // [B,NCH,D]
  __bf16* WqkvT   = (__bf16*)(ws + 181 * MB);    // [192,1024]
  __bf16* WoT     = (__bf16*)(ws + 182 * MB);    // [1024,64]
  __bf16* Wff1T   = (__bf16*)(ws + 183 * MB);    // [2048,1024]
  __bf16* Wff2T   = (__bf16*)(ws + 187 * MB);    // [1024,2048]
  __bf16* hidden  = (__bf16*)bufU;               // [BL,2048] bf16 (64 MB, after u consumed)

  // 0) weight conversion + transpose to [N,K] bf16
  auto cvt = [&](const float* in, __bf16* o, int R, int C) {
    k_convert_transpose<<<(R * C + 255) / 256, 256, 0, stream>>>(in, o, R, C);
  };
  cvt(Wq, WqkvT, DD, AA);                // rows 0..63
  cvt(Wk, WqkvT + 64 * DD, DD, AA);      // rows 64..127
  cvt(Wv, WqkvT + 128 * DD, DD, AA);     // rows 128..191
  cvt(Wo, WoT, AA, DD);
  cvt(ff_w1, Wff1T, DD, FFH);
  cvt(ff_w2, Wff2T, FFH, DD);

  // 1) u = rmsnorm(x)*b_w
  k_rmsnorm_mul<<<BL, 256, 0, stream>>>(x, norm1_w, b_w, bufU);

  // 2-4) chunked SSM scan, fused residual + bf16 downcast
  k_scan_local<<<BB * NCH * (DD / 256), 256, 0, stream>>>(bufU, a_raw, bufX, cfin);
  k_scan_carry<<<(BB * DD) / 256, 256, 0, stream>>>(a_raw, cfin, cin);
  k_scan_fixup<<<BB * NCH * (DD / 256), 256, 0, stream>>>(x, a_raw, c_w, cin, bufX, bufXbf);

  // 5) gate
  k_gate<<<BL / 8, 256, 0, stream>>>(bufX, Wg, gbias, gate);

  // 6) fused QKV projection: [BL,1024] x [1024,192]   (WN=2 -> block 128x64)
  k_gemm_bf16<0, 2><<<dim3(192 / 64, BL / 128), 256, 0, stream>>>(
      bufXbf, WqkvT, qkv, nullptr, nullptr, nullptr, 192, DD);

  // 7) sliding-window attention
  k_swa<<<BL / 8, 256, 0, stream>>>(qkv, attn, attnbf);

  // 8) Wo projection + gated residual (in place on bufX): x2 = x_ssm + gate*attn@Wo
  k_gemm_bf16<2, 4><<<dim3(DD / 128, BL / 128), 256, 0, stream>>>(
      attnbf, WoT, bufX, nullptr, bufX, gate, DD, AA);

  // 9) n2 = rmsnorm(x2)*norm2_w -> bf16
  k_rmsnorm_bf16<<<BL, 256, 0, stream>>>(bufX, norm2_w, bufXbf);

  // 10) FFN up + silu -> bf16 hidden: [BL,1024] x [1024,2048]  (WN=4 -> block 128x128)
  k_gemm_bf16<1, 4><<<dim3(FFH / 128, BL / 128), 256, 0, stream>>>(
      bufXbf, Wff1T, nullptr, hidden, nullptr, nullptr, FFH, DD);

  // 11) FFN down + residual: y = x2 + hidden x [2048,1024]
  k_gemm_bf16<3, 4><<<dim3(DD / 128, BL / 128), 256, 0, stream>>>(
      hidden, Wff2T, out, nullptr, bufX, nullptr, DD, FFH);

  (void)in_sizes; (void)n_in; (void)out_size; (void)ws_size;
}
